// Model_69715909149281
// MI455X (gfx1250) — matrix operand order repaired and checked by reading, not compiled, not run on hardware
//
// ============================================================================
// CDNA5 / gfx1250 (MI455X) implementation of the FEDformer-style reference.
//
// Roofline: ~0.5 TFLOP of dense GEMM (512x512 projections, 512x2048 FFNs,
// gaussian prior chain) over 6K-9K token rows; largest activation 75MB fits
// the 192MB L2, so compute-bound -> all linears on v_wmma_f32_16x16x32_bf16
// (f32 accumulate). B panels are staged global->LDS with the CDNA5 async
// data path (global_load_async_to_lds_b128 + s_wait_asynccnt). Each wave
// computes a 16x64 strip (4 WMMA accumulators reusing one A fragment); a
// 256-thread block covers a 128x64 macro-tile. Fourier/DFT pieces are
// O(GFLOP) and stay on VALU.
// ============================================================================
#include <hip/hip_runtime.h>
#include <math.h>

typedef __attribute__((ext_vector_type(16))) __bf16 v16bf;
typedef __attribute__((ext_vector_type(8)))  float  v8f;

#define PI_F 3.14159265358979f

// ---------------------------------------------------------------------------
// WMMA GEMM: C[M,N] = act(A[M,K] @ B[K,N] + bias + resid)
// grid = (N/64, M/128), block = 256 (8 waves; wave w owns rows m0+16w..+15,
// and 4 column tiles n0..n0+63). act: 0=none, 1=gelu(erf), 2=tanh
// ---------------------------------------------------------------------------
__global__ __launch_bounds__(256)
void gemm_wmma_kernel(const float* __restrict__ A, int lda,
                      const float* __restrict__ Bw, int ldb,
                      const float* __restrict__ bias,
                      const float* __restrict__ resid, int ldr,
                      float* __restrict__ C, int ldc,
                      int K, int act)
{
    // Braw: raw f32 B panel [k=32][n=64] filled by the async DMA path (8KB)
    // Bpk : bf16-pair panel [kpair=16][n=64], u32 = {K=2kp+1, K=2kp} (4KB)
    __shared__ __align__(16) float    Braw[32 * 64];
    __shared__ __align__(8)  unsigned Bpk[16 * 64];

    const int tid  = threadIdx.x;
    const int wave = tid >> 5;
    const int lane = tid & 31;
    const int half = lane >> 4;              // 0: lanes 0-15, 1: lanes 16-31
    const int l15  = lane & 15;
    const int n0   = blockIdx.x * 64;
    const int m0   = blockIdx.y * 128 + wave * 16;
    const int row  = m0 + l15;
    const float* arow = A + (long)row * lda;

    const v8f zero = {0.f,0.f,0.f,0.f,0.f,0.f,0.f,0.f};
    v8f accs[4] = {zero, zero, zero, zero};

    // async staging mapping: thread t copies 8 consecutive floats of row sk
    const int sk = tid >> 3;                 // 0..31 : k row of the panel
    const int sn = (tid & 7) * 8;            // col group (8 floats = 32B)
    unsigned lds_dst = (unsigned)(unsigned long long)(const void*)&Braw[sk * 64 + sn];
    // packing mapping: thread t packs kpairs {pk, pk+4, pk+8, pk+12} at col pn
    const int pk = tid >> 6;                 // 0..3
    const int pn = tid & 63;

    for (int k0 = 0; k0 < K; k0 += 32) {
        __syncthreads();   // previous iteration's fragment reads are done
        {
            unsigned long long gsrc =
                (unsigned long long)(const void*)(Bw + (long)(k0 + sk) * ldb + n0 + sn);
            asm volatile("global_load_async_to_lds_b128 %0, %1, off"
                         :: "v"(lds_dst), "v"(gsrc) : "memory");
            asm volatile("global_load_async_to_lds_b128 %0, %1, off offset:16"
                         :: "v"(lds_dst), "v"(gsrc) : "memory");
            asm volatile("s_wait_asynccnt 0" ::: "memory");
        }
        __syncthreads();   // all waves' async copies landed in Braw
        #pragma unroll
        for (int g = 0; g < 4; ++g) {
            int kp = pk + g * 4;
            __bf16 blo = (__bf16)Braw[(2 * kp) * 64 + pn];
            __bf16 bhi = (__bf16)Braw[(2 * kp + 1) * 64 + pn];
            unsigned ulo = *(const unsigned short*)&blo;
            unsigned uhi = *(const unsigned short*)&bhi;
            Bpk[kp * 64 + pn] = (uhi << 16) | ulo;
        }
        __syncthreads();   // Bpk ready
        if (k0 + 32 < K) __builtin_prefetch(arow + k0 + 32, 0, 1);

        // A fragment (16-bit A 16x32, ISA 7.12.2): lanes 0-15 hold K 0..7 and
        // 16..23, lanes 16-31 hold K 8..15 and 24..31 (2 values per VGPR).
        v16bf af;
        const float* ap = arow + k0 + half * 8;
        #pragma unroll
        for (int v = 0; v < 8; ++v) {
            int kk = (v < 4) ? (2 * v) : (16 + 2 * (v - 4));
            af[2 * v]     = (__bf16)ap[kk];
            af[2 * v + 1] = (__bf16)ap[kk + 1];
        }
        // 4 column tiles share this A fragment
        #pragma unroll
        for (int nt = 0; nt < 4; ++nt) {
            union { unsigned u[8]; v16bf v; } bf_;
            const int nn = nt * 16 + l15;
            #pragma unroll
            for (int v = 0; v < 8; ++v)
                bf_.u[v] = Bpk[(half * 4 + v + (v & 4)) * 64 + nn];   // K = half*8 + 2v (+1) for v<4, 16 + half*8 + 2(v-4) (+1) for v>=4
            accs[nt] = __builtin_amdgcn_wmma_f32_16x16x32_bf16(
                false, af, false, bf_.v, (short)0, accs[nt], false, false);
        }
    }

    // C layout: VGPR r -> M = m0 + half*8 + r, N = n0 + nt*16 + (lane&15)
    #pragma unroll
    for (int nt = 0; nt < 4; ++nt) {
        const int col = n0 + nt * 16 + l15;
        const float bb = bias ? bias[col] : 0.f;
        #pragma unroll
        for (int r = 0; r < 8; ++r) {
            int rr = m0 + half * 8 + r;
            float v = accs[nt][r] + bb;
            if (resid) v += resid[(long)rr * ldr + col];
            if (act == 1)      v = 0.5f * v * (1.f + erff(v * 0.70710678f));
            else if (act == 2) v = tanhf(v);
            C[(long)rr * ldc + col] = v;
        }
    }
}

// ---------------------------------------------------------------------------
// Build encoder embedding inputs: concat(x_enc, x_dec[:, -96:]) and marks.
// ---------------------------------------------------------------------------
__global__ void build_enc_inputs_kernel(const float* __restrict__ xe,
                                        const float* __restrict__ me,
                                        const float* __restrict__ xd,
                                        const float* __restrict__ md,
                                        float* __restrict__ ein,
                                        float* __restrict__ mrk)
{
    int i = blockIdx.x * blockDim.x + threadIdx.x;
    if (i >= 64 * 192) return;
    int l = i % 192, b = i / 192;
    for (int c = 0; c < 7; ++c)
        ein[(long)i * 7 + c] = (l < 96) ? xe[((long)b * 96 + l) * 7 + c]
                                        : xd[((long)b * 144 + (l - 48)) * 7 + c];
    for (int j = 0; j < 4; ++j)
        mrk[(long)i * 4 + j] = (l < 96) ? me[((long)b * 96 + l) * 4 + j]
                                        : md[((long)b * 144 + (l - 48)) * 4 + j];
}

// circ_conv3(x, conv) + mark @ timew : x [64,L,Cin] -> out [64,L,512]
__global__ void embed_kernel(const float* __restrict__ x, int L, int Cin,
                             const float* __restrict__ conv,
                             const float* __restrict__ mark, int Cm,
                             const float* __restrict__ timew,
                             float* __restrict__ out)
{
    long i = (long)blockIdx.x * blockDim.x + threadIdx.x;
    long total = (long)64 * L * 512;
    if (i >= total) return;
    int d = i % 512; long t = i / 512; int l = t % L; int b = (int)(t / L);
    int lm = (l + L - 1) % L, lp = (l + 1) % L;
    float acc = 0.f;
    for (int c = 0; c < Cin; ++c) {
        const float* cw = conv + ((long)d * Cin + c) * 3;
        acc += x[((long)b * L + lm) * Cin + c] * cw[0]
             + x[((long)b * L + l ) * Cin + c] * cw[1]
             + x[((long)b * L + lp) * Cin + c] * cw[2];
    }
    for (int j = 0; j < Cm; ++j)
        acc += mark[((long)b * L + l) * Cm + j] * timew[(long)j * 512 + d];
    out[i] = acc;
}

// dst [64,96,512] = src[:, l0:l0+96, :] of [64,192,512]
__global__ void slice_time_kernel(const float* __restrict__ src, int l0,
                                  float* __restrict__ dst)
{
    long i = (long)blockIdx.x * blockDim.x + threadIdx.x;
    if (i >= (long)6144 * 512) return;
    int d = i % 512; long t = i / 512; int l = t % 96; int b = (int)(t / 96);
    dst[i] = src[(((long)b * 192 + l0 + l) * 512) + d];
}

// series decomposition: x = a (+ b); trend = movmean(x, 25, replicate pad);
// seas = x - trend. tmode: 0=skip trend, 1=store trend, 2=accumulate trend.
__global__ void decomp_kernel(const float* __restrict__ a,
                              const float* __restrict__ b,
                              float* __restrict__ seas,
                              float* __restrict__ trend, int tmode,
                              int Bn, int L, int Dn)
{
    long i = (long)blockIdx.x * blockDim.x + threadIdx.x;
    long total = (long)Bn * L * Dn;
    if (i >= total) return;
    int d = i % Dn; long t = i / Dn; int l = t % L; int bb = (int)(t / L);
    const int p = (25 - 1) / 2;
    float s = 0.f;
    for (int j = -p; j <= p; ++j) {
        int lj = l + j; lj = lj < 0 ? 0 : (lj >= L ? L - 1 : lj);
        long idx = ((long)bb * L + lj) * Dn + d;
        float v = a[idx]; if (b) v += b[idx];
        s += v;
    }
    float m = s * (1.f / 25.f);
    long i0 = ((long)bb * L + l) * Dn + d;
    float x0 = a[i0] + (b ? b[i0] : 0.f);
    seas[i0] = x0 - m;
    if (trend) { if (tmode == 2) trend[i0] += m; else trend[i0] = m; }
}

// per-token layernorm over D=512, scale g, shift b. grid = tokens, block = 256
__global__ void layernorm_kernel(const float* __restrict__ x,
                                 const float* __restrict__ g,
                                 const float* __restrict__ bta,
                                 float* __restrict__ out)
{
    __shared__ float red[256];
    long tok = blockIdx.x;
    const float* xp = x + tok * 512;
    float s = 0.f;
    for (int i = threadIdx.x; i < 512; i += 256) s += xp[i];
    red[threadIdx.x] = s; __syncthreads();
    for (int o = 128; o > 0; o >>= 1) { if (threadIdx.x < o) red[threadIdx.x] += red[threadIdx.x + o]; __syncthreads(); }
    float mu = red[0] * (1.f / 512.f); __syncthreads();
    float v = 0.f;
    for (int i = threadIdx.x; i < 512; i += 256) { float d0 = xp[i] - mu; v += d0 * d0; }
    red[threadIdx.x] = v; __syncthreads();
    for (int o = 128; o > 0; o >>= 1) { if (threadIdx.x < o) red[threadIdx.x] += red[threadIdx.x + o]; __syncthreads(); }
    float inv = rsqrtf(red[0] * (1.f / 512.f) + 1e-5f);
    for (int i = threadIdx.x; i < 512; i += 256)
        out[tok * 512 + i] = (xp[i] - mu) * inv * g[i] + bta[i];
}

// subtract mean over time axis L: one thread per (b,d)
__global__ void sub_time_mean_kernel(float* __restrict__ x, int L)
{
    int i = blockIdx.x * blockDim.x + threadIdx.x;
    if (i >= 64 * 512) return;
    int d = i % 512, b = i / 512;
    float s = 0.f;
    for (int l = 0; l < L; ++l) s += x[((long)b * L + l) * 512 + d];
    s /= (float)L;
    for (int l = 0; l < L; ++l) x[((long)b * L + l) * 512 + d] -= s;
}

// rfft low modes: X [64,L,512] read as [B,L,H,E] -> xr/xi [B,H,E,32]
__global__ void dft_fwd_kernel(const float* __restrict__ X,
                               float* __restrict__ xr, float* __restrict__ xi,
                               int L)
{
    int i = blockIdx.x * blockDim.x + threadIdx.x;
    if (i >= 1048576) return;                    // 64*8*64*32
    int m = i & 31, e = (i >> 5) & 63, h = (i >> 11) & 7, b = i >> 14;
    float w = -2.f * PI_F * (float)m / (float)L;
    float sr = 0.f, si = 0.f;
    const float* xp = X + (long)b * L * 512 + h * 64 + e;
    for (int l = 0; l < L; ++l) {
        float v = xp[(long)l * 512];
        float ang = w * (float)l;
        sr += v * cosf(ang); si += v * sinf(ang);
    }
    xr[i] = sr; xi[i] = si;
}

// out[b,h,o,m] = sum_e x[b,h,e,m] * (wr + i wi)[h,e,o,m]
__global__ void mode_mix_kernel(const float* __restrict__ xr, const float* __restrict__ xi,
                                const float* __restrict__ wr, const float* __restrict__ wi,
                                float* __restrict__ outr, float* __restrict__ outi)
{
    int i = blockIdx.x * blockDim.x + threadIdx.x;
    if (i >= 1048576) return;
    int m = i & 31, o = (i >> 5) & 63, h = (i >> 11) & 7, b = i >> 14;
    float ar = 0.f, ai = 0.f;
    for (int e = 0; e < 64; ++e) {
        int xidx = ((b * 8 + h) * 64 + e) * 32 + m;
        int widx = ((h * 64 + e) * 64 + o) * 32 + m;
        float qr = xr[xidx], qi = xi[xidx];
        float cr = wr[widx], ci = wi[widx];
        ar += qr * cr - qi * ci;
        ai += qr * ci + qi * cr;
    }
    outr[i] = ar; outi[i] = ai;
}

// irfft of low-mode spectrum, natural flat order [B,H,E,L] (the reshape "bug"
// means the next GEMM reads this buffer linearly as [B*L, 512]).
__global__ void idft_kernel(const float* __restrict__ sr, const float* __restrict__ si,
                            float* __restrict__ out, int L, float scale)
{
    long i = (long)blockIdx.x * blockDim.x + threadIdx.x;
    long total = (long)32768 * L;                // 64*8*64*L
    if (i >= total) return;
    int l = (int)(i % L); int t = (int)(i / L);
    int base = t * 32;
    float acc = sr[base];                         // mode 0 (imag ignored, numpy irfft)
    float w0 = 2.f * PI_F * (float)l / (float)L;
    for (int m = 1; m < 32; ++m) {
        float ang = w0 * (float)m;
        acc += 2.f * (sr[base + m] * cosf(ang) - si[base + m] * sinf(ang));
    }
    out[i] = acc * scale / (float)L;
}

// complex scores s[b,h,x,y] = tanh_c( sum_e q[b,h,e,x]*k[b,h,e,y] )
__global__ void cross_scores_kernel(const float* __restrict__ qr, const float* __restrict__ qi,
                                    const float* __restrict__ kr, const float* __restrict__ ki,
                                    float* __restrict__ tr, float* __restrict__ ti)
{
    int i = blockIdx.x * blockDim.x + threadIdx.x;
    if (i >= 524288) return;                     // 64*8*32*32
    int y = i & 31, x = (i >> 5) & 31, h = (i >> 10) & 7, b = i >> 13;
    float ar = 0.f, ai = 0.f;
    for (int e = 0; e < 64; ++e) {
        int qidx = ((b * 8 + h) * 64 + e) * 32 + x;
        int kidx = ((b * 8 + h) * 64 + e) * 32 + y;
        float a = qr[qidx], bb = qi[qidx], c = kr[kidx], d0 = ki[kidx];
        ar += a * c - bb * d0;
        ai += a * d0 + bb * c;
    }
    // complex tanh(a+ib) = (sinh 2a + i sin 2b) / (cosh 2a + cos 2b)
    float a2 = 2.f * ar, b2 = 2.f * ai, outr, outi;
    if (fabsf(a2) > 40.f) { outr = a2 > 0.f ? 1.f : -1.f; outi = 0.f; }
    else {
        float den = coshf(a2) + cosf(b2);
        den = (fabsf(den) < 1e-20f) ? 1e-20f : den;
        outr = sinhf(a2) / den; outi = sinf(b2) / den;
    }
    tr[i] = outr; ti[i] = outi;
}

// v[b,h,e,x] = sum_y s[b,h,x,y] * k[b,h,e,y]   (complex)
__global__ void cross_apply_kernel(const float* __restrict__ tr, const float* __restrict__ ti,
                                   const float* __restrict__ kr, const float* __restrict__ ki,
                                   float* __restrict__ vr, float* __restrict__ vi)
{
    int i = blockIdx.x * blockDim.x + threadIdx.x;
    if (i >= 1048576) return;                    // 64*8*64*32
    int x = i & 31, e = (i >> 5) & 63, h = (i >> 11) & 7, b = i >> 14;
    float ar = 0.f, ai = 0.f;
    for (int y = 0; y < 32; ++y) {
        int tidx = ((b * 8 + h) * 32 + x) * 32 + y;
        int kidx = ((b * 8 + h) * 64 + e) * 32 + y;
        float a = tr[tidx], bb = ti[tidx], c = kr[kidx], d0 = ki[kidx];
        ar += a * c - bb * d0;
        ai += a * d0 + bb * c;
    }
    vr[i] = ar; vi[i] = ai;
}

__global__ void add_kernel(const float* __restrict__ a, const float* __restrict__ b,
                           float* __restrict__ c, int n)
{
    int i = blockIdx.x * blockDim.x + threadIdx.x;
    if (i < n) c[i] = a[i] + b[i];
}

__global__ void mean_xenc_kernel(const float* __restrict__ xe, float* __restrict__ mean)
{
    int i = blockIdx.x * blockDim.x + threadIdx.x;
    if (i >= 64 * 7) return;
    int c = i % 7, b = i / 7;
    float s = 0.f;
    for (int l = 0; l < 96; ++l) s += xe[((long)b * 96 + l) * 7 + c];
    mean[i] = s * (1.f / 96.f);
}

// build decoder seasonal/trend init buffers [64,144,7]
__global__ void build_dec_kernel(const float* __restrict__ seasi,
                                 const float* __restrict__ trendi,
                                 const float* __restrict__ mean,
                                 float* __restrict__ seasD, float* __restrict__ trendD)
{
    int i = blockIdx.x * blockDim.x + threadIdx.x;
    if (i >= 64 * 144 * 7) return;
    int c = i % 7; int t = i / 7; int l = t % 144; int b = t / 144;
    float sv, tv;
    if (l < 48) { sv = seasi[((long)b * 96 + 48 + l) * 7 + c];
                  tv = trendi[((long)b * 96 + 48 + l) * 7 + c]; }
    else        { sv = 0.f; tv = mean[b * 7 + c]; }
    seasD[i] = sv; trendD[i] = tv;
}

// trendD += circ_conv3(tsum [64,144,512], tp [7,512,3])
__global__ void trend_conv_kernel(const float* __restrict__ tsum,
                                  const float* __restrict__ tp,
                                  float* __restrict__ trendD)
{
    int i = blockIdx.x * blockDim.x + threadIdx.x;
    if (i >= 64 * 144 * 7) return;
    int o = i % 7; int t = i / 7; int l = t % 144; int b = t / 144;
    int lm = (l + 143) % 144, lp = (l + 1) % 144;
    float acc = 0.f;
    for (int d = 0; d < 512; ++d) {
        const float* w = tp + ((long)o * 512 + d) * 3;
        acc += tsum[((long)b * 144 + lm) * 512 + d] * w[0]
             + tsum[((long)b * 144 + l ) * 512 + d] * w[1]
             + tsum[((long)b * 144 + lp) * 512 + d] * w[2];
    }
    trendD[i] += acc;
}

// dec_out[:, -96:] = (xln @ dec_proj.w + b) + trend  -> d_out[0:43008]
__global__ void final_kernel(const float* __restrict__ xln,
                             const float* __restrict__ pw, const float* __restrict__ pb,
                             const float* __restrict__ trendD, float* __restrict__ dout)
{
    int i = blockIdx.x * blockDim.x + threadIdx.x;
    if (i >= 64 * 96 * 7) return;
    int o = i % 7; int t = i / 7; int lo = t % 96; int b = t / 96;
    int l = lo + 48;
    float acc = pb[o];
    for (int d = 0; d < 512; ++d)
        acc += xln[((long)b * 144 + l) * 512 + d] * pw[(long)d * 7 + o];
    dout[i] = acc + trendD[((long)b * 144 + l) * 7 + o];
}

// KL partials (deterministic two-stage reduction, 256 fixed blocks)
__global__ void kl_partial_kernel(const float* __restrict__ pp,
                                  const float* __restrict__ qp,
                                  float* __restrict__ partial)
{
    __shared__ float red[256];
    const long total = (long)6144 * 512;
    float local = 0.f;
    for (long i = (long)blockIdx.x * blockDim.x + threadIdx.x; i < total;
         i += (long)gridDim.x * blockDim.x) {
        int d = (int)(i % 512); long r = i / 512;
        float pm  = pp[r * 1024 + d];
        float qm  = qp[r * 1024 + d];
        float qlv = qp[r * 1024 + 512 + d];
        local += 1.f + (qlv - pm) - ((pm - qm) * (pm - qm) + expf(qlv)) / expf(pm);
    }
    red[threadIdx.x] = local; __syncthreads();
    for (int o = 128; o > 0; o >>= 1) { if (threadIdx.x < o) red[threadIdx.x] += red[threadIdx.x + o]; __syncthreads(); }
    if (threadIdx.x == 0) partial[blockIdx.x] = red[0];
}

__global__ void kl_final_kernel(const float* __restrict__ partial, float* __restrict__ out)
{
    __shared__ float red[256];
    red[threadIdx.x] = partial[threadIdx.x]; __syncthreads();
    for (int o = 128; o > 0; o >>= 1) { if (threadIdx.x < o) red[threadIdx.x] += red[threadIdx.x + o]; __syncthreads(); }
    if (threadIdx.x == 0) *out = -0.5f * red[0];
}

// ===========================================================================
extern "C" void kernel_launch(void* const* d_in, const int* in_sizes, int n_in,
                              void* d_out_v, int out_size, void* d_ws, size_t ws_size,
                              hipStream_t stream)
{
    (void)in_sizes; (void)n_in; (void)out_size; (void)ws_size;
    auto PRM = [&](int i) { return (const float*)d_in[i]; };
    // jax pytree flatten order (dict keys sorted):
    // 0-3 inputs; 4 cross_fb.wi 5 cross_fb.wr; 6 dec_emb.conv 7 dec_emb.time;
    // 8 dl.conv1 9 dl.conv2; 10-17 dl.cross_attn{k.b,k.w,o.b,o.w,q.b,q.w,v.b,v.w};
    // 18-25 dl.self_attn{...}; 26 dl.self_fb.wi 27 .wr; 28 dl.trend_proj;
    // 29 dec_norm.b 30 .g; 31 dec_proj.b 32 .w; 33 enc_emb.conv 34 .time;
    // 35+12*li enc_layers[li]{attn k.b,k.w,o.b,o.w,q.b,q.w,v.b,v.w, conv1,conv2, fb.wi,fb.wr};
    // 59 enc_norm.b 60 .g; 61-68 gauss.attn{...}; 69 post.b 70 post.w;
    // 71 prior1.b 72 .w; 73 prior2.b 74 .w; 75 rec1.b 76 .w; 77 rec2.b 78 .w
    const float* x_enc = PRM(0);
    const float* m_enc = PRM(1);
    const float* x_dec = PRM(2);
    const float* m_dec = PRM(3);
    float* dout = (float*)d_out_v;

    float* ws = (float*)d_ws;
    size_t off = 0;
    auto alloc = [&](size_t n) -> float* {
        float* p = ws + off; off += (n + 63) & ~(size_t)63; return p;
    };
    float* EMB   = alloc((size_t)64 * 192 * 512);
    float* EIN   = alloc((size_t)64 * 192 * 7);
    float* MRK   = alloc((size_t)64 * 192 * 4);
    float* XA    = alloc((size_t)9216 * 512);
    float* XB    = alloc((size_t)9216 * 512);
    float* T1    = alloc((size_t)9216 * 512);
    float* T2    = alloc((size_t)9216 * 1024);   // also rec hidden [6144,1024]
    float* T3    = alloc((size_t)9216 * 512);
    float* TS    = alloc((size_t)9216 * 512);
    float* FF    = alloc((size_t)9216 * 2048);
    float* ENC   = alloc((size_t)6144 * 512);
    float* POST  = alloc((size_t)6144 * 512);
    float* ENC2  = alloc((size_t)6144 * 512);
    float* PP    = alloc((size_t)6144 * 1024);
    float* QP    = alloc((size_t)6144 * 1024);
    float* F0 = alloc(1048576); float* F1 = alloc(1048576);
    float* F2 = alloc(1048576); float* F3 = alloc(1048576);
    float* F4 = alloc(1048576); float* F5 = alloc(1048576);
    float* F6 = alloc(1048576); float* F7 = alloc(1048576);
    float* SEASI  = alloc(43008);
    float* TRENDI = alloc(43008);
    float* MEANB  = alloc(448);
    float* SEASD  = alloc(64512);
    float* TRENDD = alloc(64512);
    float* KLP    = alloc(256);

    auto gemm = [&](const float* A, int lda, const float* Bw, int ldb,
                    const float* bias, const float* resid, int ldr,
                    float* C, int ldc, int M, int N, int K, int act) {
        dim3 g(N / 64, M / 128);
        gemm_wmma_kernel<<<g, 256, 0, stream>>>(A, lda, Bw, ldb, bias, resid, ldr, C, ldc, K, act);
    };
    auto dft = [&](const float* X, float* xr, float* xi, int L) {
        dft_fwd_kernel<<<1048576 / 256, 256, 0, stream>>>(X, xr, xi, L);
    };
    auto mix = [&](const float* xr, const float* xi, const float* wr, const float* wi,
                   float* outr, float* outi) {
        mode_mix_kernel<<<1048576 / 256, 256, 0, stream>>>(xr, xi, wr, wi, outr, outi);
    };
    auto idft = [&](const float* sr, const float* si, float* out, int L, float scale) {
        long total = (long)32768 * L;
        idft_kernel<<<(int)((total + 255) / 256), 256, 0, stream>>>(sr, si, out, L, scale);
    };
    auto decompL = [&](const float* a, const float* b, float* seas, float* trend,
                       int tmode, int Bn, int L, int Dn) {
        long total = (long)Bn * L * Dn;
        decomp_kernel<<<(int)((total + 255) / 256), 256, 0, stream>>>(a, b, seas, trend, tmode, Bn, L, Dn);
    };
    auto lnorm = [&](const float* x, const float* g, const float* b, float* out, int tokens) {
        layernorm_kernel<<<tokens, 256, 0, stream>>>(x, g, b, out);
    };
    auto subtime = [&](float* x, int L) {
        sub_time_mean_kernel<<<(64 * 512 + 255) / 256, 256, 0, stream>>>(x, L);
    };

    // ---- embeddings -------------------------------------------------------
    build_enc_inputs_kernel<<<(64 * 192 + 255) / 256, 256, 0, stream>>>(
        x_enc, m_enc, x_dec, m_dec, EIN, MRK);
    embed_kernel<<<(64 * 192 * 512) / 256, 256, 0, stream>>>(
        EIN, 192, 7, PRM(33), MRK, 4, PRM(34), EMB);

    // ---- encoder (run twice: [:, :96] -> ENC, [:, 96:] -> POST) -----------
    auto run_enc_pass = [&](int l0, float* result) {
        slice_time_kernel<<<((long)6144 * 512) / 256, 256, 0, stream>>>(EMB, l0, XA);
        float* X = XA; float* Y = XB;
        for (int li = 0; li < 2; ++li) {
            int bs = 35 + li * 12;
            gemm(X, 512, PRM(bs + 5), 512, PRM(bs + 4), nullptr, 0, T1, 512, 6144, 512, 512, 0);
            dft(T1, F0, F1, 96);
            mix(F0, F1, PRM(bs + 11), PRM(bs + 10), F2, F3);
            idft(F2, F3, T3, 96, 1.f);
            gemm(T3, 512, PRM(bs + 3), 512, PRM(bs + 2), nullptr, 0, T1, 512, 6144, 512, 512, 0);
            decompL(X, T1, Y, nullptr, 0, 64, 96, 512); { float* t = X; X = Y; Y = t; }
            gemm(X, 512, PRM(bs + 8), 2048, nullptr, nullptr, 0, FF, 2048, 6144, 2048, 512, 1);
            gemm(FF, 2048, PRM(bs + 9), 512, nullptr, nullptr, 0, T1, 512, 6144, 512, 2048, 0);
            decompL(X, T1, Y, nullptr, 0, 64, 96, 512); { float* t = X; X = Y; Y = t; }
        }
        lnorm(X, PRM(60), PRM(59), result, 6144);
        subtime(result, 96);
    };
    run_enc_pass(0, ENC);
    run_enc_pass(96, POST);

    // ---- fourier cross attention helper (K always 6144 enc tokens) --------
    auto fourier_cross_attn = [&](const float* Qx, int Lq, const float* Kx,
                                  const float* qw, const float* qb,
                                  const float* kw, const float* kb,
                                  const float* ow, const float* ob,
                                  const float* resid, float* Cout) {
        int Mq = 64 * Lq;
        gemm(Qx, 512, qw, 512, qb, nullptr, 0, T1, 512, Mq, 512, 512, 0);
        gemm(Kx, 512, kw, 512, kb, nullptr, 0, T3, 512, 6144, 512, 512, 0);
        dft(T1, F0, F1, Lq);
        dft(T3, F2, F3, 96);
        cross_scores_kernel<<<524288 / 256, 256, 0, stream>>>(F0, F1, F2, F3, F4, F5);
        cross_apply_kernel<<<1048576 / 256, 256, 0, stream>>>(F4, F5, F2, F3, F6, F7);
        mix(F6, F7, PRM(5), PRM(4), F0, F1);
        idft(F0, F1, T1, Lq, 1.f / (512.f * 512.f));
        gemm(T1, 512, ow, 512, ob, resid, 512, Cout, 512, Mq, 512, 512, 0);
    };

    // ---- gaussian net -----------------------------------------------------
    fourier_cross_attn(ENC, 96, ENC, PRM(66), PRM(65), PRM(62), PRM(61),
                       PRM(64), PRM(63), nullptr, XA);
    gemm(XA, 512, PRM(72), 2048, PRM(71), nullptr, 0, FF, 2048, 6144, 2048, 512, 2); // tanh
    gemm(FF, 2048, PRM(74), 1024, PRM(73), nullptr, 0, PP, 1024, 6144, 1024, 2048, 0);
    gemm(PP, 1024, PRM(76), 1024, PRM(75), nullptr, 0, T2, 1024, 6144, 1024, 512, 1); // gelu(pm@rec1)
    gemm(T2, 1024, PRM(78), 512, PRM(77), nullptr, 0, T3, 512, 6144, 512, 1024, 0);
    add_kernel<<<(3145728 + 255) / 256, 256, 0, stream>>>(ENC, T3, ENC2, 3145728);
    fourier_cross_attn(POST, 96, ENC, PRM(66), PRM(65), PRM(62), PRM(61),
                       PRM(64), PRM(63), nullptr, XA);
    gemm(XA, 512, PRM(70), 1024, PRM(69), nullptr, 0, QP, 1024, 6144, 1024, 512, 0);

    // ---- decoder prep -----------------------------------------------------
    decompL(x_enc, nullptr, SEASI, TRENDI, 1, 64, 96, 7);
    mean_xenc_kernel<<<2, 256, 0, stream>>>(x_enc, MEANB);
    build_dec_kernel<<<(64512 + 255) / 256, 256, 0, stream>>>(SEASI, TRENDI, MEANB, SEASD, TRENDD);
    embed_kernel<<<(64 * 144 * 512) / 256, 256, 0, stream>>>(
        SEASD, 144, 7, PRM(6), m_dec, 4, PRM(7), XA);

    // ---- decoder layer ----------------------------------------------------
    // self attention (fourier block) with residual folded into O-projection
    gemm(XA, 512, PRM(23), 512, PRM(22), nullptr, 0, T1, 512, 9216, 512, 512, 0);
    dft(T1, F0, F1, 144);
    mix(F0, F1, PRM(27), PRM(26), F2, F3);
    idft(F2, F3, T1, 144, 1.f);
    gemm(T1, 512, PRM(21), 512, PRM(20), XA, 512, XB, 512, 9216, 512, 512, 0);
    decompL(XB, nullptr, XA, TS, 1, 64, 144, 512);                // x -> XA, t1 -> TS
    // cross attention against ENC2
    fourier_cross_attn(XA, 144, ENC2, PRM(15), PRM(14), PRM(11), PRM(10),
                       PRM(13), PRM(12), XA, XB);
    decompL(XB, nullptr, XA, TS, 2, 64, 144, 512);                // t2 += trend
    // FFN
    gemm(XA, 512, PRM(8), 2048, nullptr, nullptr, 0, FF, 2048, 9216, 2048, 512, 1);
    gemm(FF, 2048, PRM(9), 512, nullptr, nullptr, 0, T1, 512, 9216, 512, 2048, 0);
    decompL(XA, T1, XB, TS, 2, 64, 144, 512);                     // t3 += trend, x -> XB
    trend_conv_kernel<<<(64512 + 255) / 256, 256, 0, stream>>>(TS, PRM(28), TRENDD);

    // ---- decoder head -----------------------------------------------------
    lnorm(XB, PRM(30), PRM(29), XA, 9216);
    subtime(XA, 144);
    final_kernel<<<(43008 + 255) / 256, 256, 0, stream>>>(XA, PRM(32), PRM(31), TRENDD, dout);

    // ---- KL ---------------------------------------------------------------
    kl_partial_kernel<<<256, 256, 0, stream>>>(PP, QP, KLP);
    kl_final_kernel<<<1, 256, 0, stream>>>(KLP, dout + 43008);
}
